// MLRAttention_75385265979853
// MI455X (gfx1250) — matrix operand order repaired and checked by reading, not compiled, not run on hardware
//
#include <hip/hip_runtime.h>
#include <hip/hip_bf16.h>
#include <math.h>

// ---------------------------------------------------------------------------
// MLR (multi-scale low-rank) attention for MI455X / gfx1250.
//  - Dense GEMMs (QKV, output projection): bf16 inputs + fp32 accumulate via
//    V_WMMA_F32_16X16X32_BF16 (8x FLOP/instruction vs fp32 WMMA; these stages
//    carry ~54% of the 32 GFLOP total and are matrix-issue bound since all
//    data lives in the 192MB L2).
//  - Attention scores / softmax / P@V: exact fp32 via V_WMMA_F32_16X16X4_F32
//    (K=4 granule matches the hierarchical rank prefix depths 32/48/56/64).
// ---------------------------------------------------------------------------

typedef __attribute__((ext_vector_type(2)))  float  v2f;
typedef __attribute__((ext_vector_type(8)))  float  v8f;
typedef __attribute__((ext_vector_type(16))) __bf16 v16bf;

#define D_MODEL 1024
#define N_HEAD  16
#define SEQ_T   2048
#define HEAD_D  64

static __device__ __forceinline__ v8f wmma_f32(v2f a, v2f b, v8f c) {
  // D = A(16x4,f32) x B(4x16,f32) + C(16x16,f32)
  return __builtin_amdgcn_wmma_f32_16x16x4_f32(
      /*neg_a=*/false, a, /*neg_b=*/false, b,
      /*c_mod=*/(short)0, c, /*reuse_a=*/false, /*reuse_b=*/false);
}

// ---------------------------------------------------------------------------
// bf16 GEMM: C[M x N] = A[M x K] @ B[K x N] + bias[N]   (A,B fp32 in memory,
// converted to bf16 while staging into LDS; fp32 accumulation in WMMA).
// Workgroup tile 64(M) x 64(N), 4 waves; wave w computes rows [16w, 16w+16).
// Staging is split gather->convert/store so all global loads of a chunk are
// in flight together (pipelined under LOADcnt) instead of per-element waits.
// ---------------------------------------------------------------------------
__global__ __launch_bounds__(128)
void gemm_bf16_wmma_kernel(const float* __restrict__ A,
                           const float* __restrict__ B,
                           const float* __restrict__ bias,
                           float* __restrict__ C,
                           int M, int N, int K)
{
  // Row stride 36 bf16 = 72B -> 18-bank stride, conflict-free for 16 lanes.
  __shared__ __bf16 lA [64][36];   // A tile,  [m][k]
  __shared__ __bf16 lBt[64][36];   // B tile transposed, [n][k] (k contiguous)

  const int tileM = blockIdx.y * 64;
  const int tileN = blockIdx.x * 64;
  const int wave  = threadIdx.x >> 5;
  const int lane  = threadIdx.x & 31;
  const int row   = lane & 15;     // A-operand M row / B-operand N col
  const int hi    = lane >> 4;     // lane half selects K sub-range

  v8f acc[4] = {v8f{}, v8f{}, v8f{}, v8f{}};

  for (int k0 = 0; k0 < K; k0 += 32) {
    // ---- gather phase: issue ALL chunk loads back-to-back ----
    float2 va[8];                    // A: 8 float2 per thread (64x32 tile)
    float  vb0[8], vb1[8];           // B: 8 (k,k+1) pairs per thread
    #pragma unroll
    for (int it = 0; it < 8; ++it) {
      int p  = threadIdx.x + it * 128;     // 0..1023
      int r  = p >> 4;                     // 64 rows x 16 pairs
      int c2 = (p & 15) * 2;
      va[it] = *reinterpret_cast<const float2*>(
          &A[(size_t)(tileM + r) * K + (k0 + c2)]);
    }
    #pragma unroll
    for (int it = 0; it < 8; ++it) {
      int p = threadIdx.x + it * 128;      // 0..1023
      int c = p & 63;                      // column within tile (coalesced)
      int k = (p >> 6) * 2;                // 0,2,..,30
      vb0[it] = B[(size_t)(k0 + k)     * N + (tileN + c)];
      vb1[it] = B[(size_t)(k0 + k + 1) * N + (tileN + c)];
    }
    // Prefetch next K-chunk while this one is consumed (global_prefetch_b8).
    if (k0 + 32 < K) {
      __builtin_prefetch(&A[(size_t)(tileM + (threadIdx.x & 63)) * K + k0 + 32]);
      __builtin_prefetch(&B[(size_t)(k0 + 32 + (threadIdx.x & 31)) * N + tileN]);
    }
    // ---- drain phase: convert + packed bf16 stores (cvt_pk + ds_store_b32)
    #pragma unroll
    for (int it = 0; it < 8; ++it) {
      int p  = threadIdx.x + it * 128;
      int r  = p >> 4;
      int c2 = (p & 15) * 2;
      lA[r][c2]     = (__bf16)va[it].x;
      lA[r][c2 + 1] = (__bf16)va[it].y;
    }
    #pragma unroll
    for (int it = 0; it < 8; ++it) {
      int p = threadIdx.x + it * 128;
      int c = p & 63;
      int k = (p >> 6) * 2;
      lBt[c][k]     = (__bf16)vb0[it];
      lBt[c][k + 1] = (__bf16)vb1[it];
    }
    __syncthreads();

    // ---- A operand (16x32; this kernel's contiguous k order, same as B):
    // vgpr j<4 : K = 2j + 16*hi ; vgpr j>=4 : K = 8 + 2(j-4) + 16*hi
    v16bf a;
    #pragma unroll
    for (int j = 0; j < 4; ++j) {
      int ka = 2 * j + 16 * hi;
      a[2 * j]     = lA[wave * 16 + row][ka];
      a[2 * j + 1] = lA[wave * 16 + row][ka + 1];
      int kb = 8 + 2 * j + 16 * hi;
      a[8 + 2 * j]     = lA[wave * 16 + row][kb];
      a[8 + 2 * j + 1] = lA[wave * 16 + row][kb + 1];
    }

    // ---- B operand (32x16): vgpr j holds K = 2j + 16*hi, col = row ----
    #pragma unroll
    for (int nt = 0; nt < 4; ++nt) {
      v16bf b;
      #pragma unroll
      for (int j = 0; j < 8; ++j) {
        int k = 2 * j + 16 * hi;
        b[2 * j]     = lBt[nt * 16 + row][k];
        b[2 * j + 1] = lBt[nt * 16 + row][k + 1];
      }
      acc[nt] = __builtin_amdgcn_wmma_f32_16x16x32_bf16(
          false, a, false, b, (short)0, acc[nt], false, false);
    }
    __syncthreads();
  }

  // Writeback. C/D layout: lane = n + 16*(m>=8); vgpr r holds m = 8*(lane>=16)+r
  const int ncol  = lane & 15;
  const int mbase = (lane >> 4) * 8;
  #pragma unroll
  for (int nt = 0; nt < 4; ++nt) {
    int n = tileN + nt * 16 + ncol;
    float bn = bias ? bias[n] : 0.0f;
    #pragma unroll
    for (int r = 0; r < 8; ++r) {
      int m = tileM + wave * 16 + mbase + r;
      C[(size_t)m * N + n] = acc[nt][r] + bn;
    }
  }
}

// ---------------------------------------------------------------------------
// Fused MLR attention, flash-style: one wave32 per (head, 16-query-row tile).
// qkv layout: [T][3*D_MODEL] fp32 (q | k | v), head h occupies cols h*64..h*64+63.
// Score(t,s) = scaled-prefix dot product; prefix depth per 16x16 tile:
//   same 256-block: 64 | same 512: 56 | same 1024: 48 | else: 32
// (q pre-scaled per rank segment: 1/32, 1/16, 1/8, 1/8).
// ---------------------------------------------------------------------------
__global__ __launch_bounds__(32)
void mlr_attn_kernel(const float* __restrict__ qkv, float* __restrict__ Y)
{
  const int h    = blockIdx.y;
  const int t0   = blockIdx.x * 16;
  const int lane = threadIdx.x;

  const float* Qp = qkv + (size_t)h * HEAD_D;                 // row stride 3072
  const float* Kp = qkv + D_MODEL     + (size_t)h * HEAD_D;
  const float* Vp = qkv + 2 * D_MODEL + (size_t)h * HEAD_D;

  __shared__ float sQ[16][HEAD_D + 4];  // scaled Q tile, padded
  __shared__ float sP[16][17];          // P tile staging (C-layout -> A-layout)

  // Load and pre-scale Q tile (16x64), 32 lanes x 32 elements (uniform trips).
  #pragma unroll
  for (int it = 0; it < 32; ++it) {
    int i = lane + it * 32;
    int r = i >> 6, d = i & 63;
    float sc = (d < 32) ? (1.0f / 32.0f) : (d < 48) ? (1.0f / 16.0f) : (1.0f / 8.0f);
    sQ[r][d] = Qp[(size_t)(t0 + r) * (3 * D_MODEL) + d] * sc;
  }
  __syncthreads();

  const int mrowA = lane & 15;        // A-operand M row
  const int khi   = (lane >> 4) * 2;  // A/B hi-lane k offset
  const int ncol  = lane & 15;        // C-layout column
  const int mbase = (lane >> 4) * 8;  // C-layout row base

  float mrow[8], lrow[8];
  #pragma unroll
  for (int r = 0; r < 8; ++r) { mrow[r] = -INFINITY; lrow[r] = 0.0f; }
  v8f o[4] = {v8f{}, v8f{}, v8f{}, v8f{}};

  for (int s0 = 0; s0 <= t0; s0 += 16) {
    // Hierarchy level -> prefix depth (uniform across the wave).
    int kd;
    if      ((t0 >> 8)  == (s0 >> 8))  kd = 64;
    else if ((t0 >> 9)  == (s0 >> 9))  kd = 56;
    else if ((t0 >> 10) == (s0 >> 10)) kd = 48;
    else                               kd = 32;

    // ---- Score tile S = Qs @ K^T (prefix of head dim) ----
    v8f s = v8f{};
    for (int k0 = 0; k0 < kd; k0 += 4) {
      v2f a, b;
      a.x = sQ[mrowA][k0 + khi];
      a.y = sQ[mrowA][k0 + khi + 1];
      const float* kp = Kp + (size_t)(s0 + ncol) * (3 * D_MODEL) + k0 + khi;
      b.x = kp[0];
      b.y = kp[1];
      s = wmma_f32(a, b, s);
    }

    // Causal mask inside the diagonal tile: col (s0+ncol) > row (t0+m).
    if (s0 == t0) {
      #pragma unroll
      for (int r = 0; r < 8; ++r)
        if (ncol > mbase + r) s[r] = -INFINITY;
    }

    // ---- Online softmax (rows 0-7 in lanes 0-15, rows 8-15 in lanes 16-31) ----
    float p[8], alpha[8], rsum[8];
    #pragma unroll
    for (int r = 0; r < 8; ++r) {
      float v = s[r];
      #pragma unroll
      for (int off = 1; off < 16; off <<= 1)
        v = fmaxf(v, __shfl_xor(v, off, 32));     // reduce within 16-lane half
      float nm = fmaxf(mrow[r], v);
      alpha[r] = __expf(mrow[r] - nm);            // exp(-inf)=0 on first tile
      p[r] = __expf(s[r] - nm);
      float su = p[r];
      #pragma unroll
      for (int off = 1; off < 16; off <<= 1)
        su += __shfl_xor(su, off, 32);
      rsum[r] = su;
      mrow[r] = nm;
      lrow[r] = lrow[r] * alpha[r] + rsum[r];
    }
    #pragma unroll
    for (int nt = 0; nt < 4; ++nt)
      #pragma unroll
      for (int r = 0; r < 8; ++r)
        o[nt][r] *= alpha[r];

    // ---- Stage P (C-layout) to LDS, reread in A-layout for P@V ----
    __syncthreads();
    #pragma unroll
    for (int r = 0; r < 8; ++r)
      sP[mbase + r][ncol] = p[r];
    __syncthreads();

    #pragma unroll
    for (int kk = 0; kk < 16; kk += 4) {
      v2f a;
      a.x = sP[mrowA][kk + khi];
      a.y = sP[mrowA][kk + khi + 1];
      #pragma unroll
      for (int nt = 0; nt < 4; ++nt) {
        const float* vp = Vp + (size_t)(s0 + kk + khi) * (3 * D_MODEL) + nt * 16 + ncol;
        v2f b;
        b.x = vp[0];
        b.y = vp[3 * D_MODEL];   // next k row
        o[nt] = wmma_f32(a, b, o[nt]);
      }
    }
  }

  // Finalize: y = O / l, write to Y[T][D_MODEL] at head column block.
  #pragma unroll
  for (int nt = 0; nt < 4; ++nt)
    #pragma unroll
    for (int r = 0; r < 8; ++r) {
      int m = t0 + mbase + r;
      Y[(size_t)m * D_MODEL + h * HEAD_D + nt * 16 + ncol] = o[nt][r] / lrow[r];
    }
}

// ---------------------------------------------------------------------------
// Launch: QKV GEMM (bf16 WMMA) -> fused attention (fp32 WMMA) -> projection.
// d_ws: qkv (2048x3072 f32, 24MB) + y (2048x1024 f32, 8MB) = 32MB.
// ---------------------------------------------------------------------------
extern "C" void kernel_launch(void* const* d_in, const int* in_sizes, int n_in,
                              void* d_out, int out_size, void* d_ws, size_t ws_size,
                              hipStream_t stream) {
  const float* x      = (const float*)d_in[0];
  const float* W_attn = (const float*)d_in[1];
  const float* b_attn = (const float*)d_in[2];
  const float* W_proj = (const float*)d_in[3];
  const float* b_proj = (const float*)d_in[4];

  float* qkv = (float*)d_ws;                        // [2048][3072]
  float* y   = qkv + (size_t)SEQ_T * 3 * D_MODEL;   // [2048][1024]
  float* out = (float*)d_out;

  // qkv = x @ W_attn + b_attn   (2048 x 1024 x 3072)
  gemm_bf16_wmma_kernel<<<dim3((3 * D_MODEL) / 64, SEQ_T / 64), 128, 0, stream>>>(
      x, W_attn, b_attn, qkv, SEQ_T, 3 * D_MODEL, D_MODEL);

  // fused hierarchical attention per (head, 16-row tile)
  mlr_attn_kernel<<<dim3(SEQ_T / 16, N_HEAD), 32, 0, stream>>>(qkv, y);

  // out = y @ W_proj + b_proj   (2048 x 1024 x 1024)
  gemm_bf16_wmma_kernel<<<dim3(D_MODEL / 64, SEQ_T / 64), 128, 0, stream>>>(
      y, W_proj, b_proj, out, SEQ_T, D_MODEL, D_MODEL);
}
